// QuantumFeedForward_65481071401652
// MI455X (gfx1250) — matrix operand order repaired and checked by reading, not compiled, not run on hardware
//
#include <hip/hip_runtime.h>

typedef __attribute__((ext_vector_type(16))) _Float16 v16h;
typedef __attribute__((ext_vector_type(8)))  _Float16 v8h;
typedef __attribute__((ext_vector_type(4)))  _Float16 h4;
typedef __attribute__((ext_vector_type(2)))  _Float16 h2;
typedef __attribute__((ext_vector_type(8)))  float    v8f;

#define EMBED 512
#define FFN   2048
#define NQ    8
#define MT    32           // rows (M) per workgroup
#define CF    512          // FFN chunk processed per phase-1/phase-2 round
#define CFP   (CF + 8)     // padded LDS row stride (520 halves = 1040 B):
                           // bank stride 4 -> conflict-free A-fragment b128 loads,
                           // and addresses stay AFFINE in kb (no per-iter VALU swizzle)
#define NCH   (FFN / CF)   // 4 chunks

// ---- Pre-convert W2 (f32 [512][2048]) to fp16 in workspace ----
__global__ __launch_bounds__(256) void w2_cvt(const float* __restrict__ w2,
                                              _Float16* __restrict__ w2h) {
  int i = (blockIdx.x * 256 + threadIdx.x) * 4;
  float4 v = *(const float4*)(w2 + i);
  h4 o = { (_Float16)v.x, (_Float16)v.y, (_Float16)v.z, (_Float16)v.w };
  *(h4*)(w2h + i) = o;
}

// ---- Fused: z = cos(theta)*cos(x[:,:8]) -> h = relu(z W1^T + b1) (LDS, fp16)
// ----        -> out = h W2^T + b2 via v_wmma_f32_16x16x32_f16 ----
__global__ __launch_bounds__(256) void qffn(const float* __restrict__ x,
                                            const float* __restrict__ theta,
                                            const float* __restrict__ W1,
                                            const float* __restrict__ b1,
                                            const _Float16* __restrict__ w2h,
                                            const float* __restrict__ b2,
                                            float* __restrict__ out) {
  __shared__ __align__(16) _Float16 hbuf[MT * CFP];  // ~33.3 KB, padded rows
  __shared__ float zbuf[MT * NQ];                    // 1 KB

  const int tid   = threadIdx.x;
  const int lane  = tid & 31;
  const int wave  = tid >> 5;        // 0..7
  const int lhalf = lane >> 4;       // 0 or 1 (half-wave)
  const int nlane = lane & 15;
  const int rowbase = blockIdx.x * MT;

  // ---- quantum features for the 32 rows of this tile ----
  {
    int m = tid >> 3;                // 0..31
    int q = tid & 7;
    float xv = x[(size_t)(rowbase + m) * EMBED + q];
    zbuf[m * NQ + q] = __cosf(theta[q]) * __cosf(xv);
  }

  v8f acc0[4] = {}, acc1[4] = {};    // [col tile]; acc0: rows 0-15, acc1: rows 16-31

  const int floc = tid * 2;          // this thread's 2 FFN columns per chunk

  for (int ch = 0; ch < NCH; ++ch) {
    __syncthreads();                 // previous chunk's WMMA reads finished

    // ---- phase 1: h[m][ch*CF + floc, +1] for all 32 rows (scalar FMAs, K=8) ----
    {
      int fglob = ch * CF + floc;
      const float* w1p = W1 + (size_t)fglob * NQ;  // W1 is [FFN][NQ] row-major
      float wA[NQ], wB[NQ];
#pragma unroll
      for (int q = 0; q < NQ; ++q) { wA[q] = w1p[q]; wB[q] = w1p[NQ + q]; }
      float bA = b1[fglob], bB = b1[fglob + 1];
#pragma unroll
      for (int m = 0; m < MT; ++m) {
        float a0 = bA, a1 = bB;
#pragma unroll
        for (int q = 0; q < NQ; ++q) {
          float zz = zbuf[m * NQ + q];
          a0 = fmaf(zz, wA[q], a0);
          a1 = fmaf(zz, wB[q], a1);
        }
        a0 = fmaxf(a0, 0.0f);
        a1 = fmaxf(a1, 0.0f);
        h2 p = { (_Float16)a0, (_Float16)a1 };
        *(h2*)&hbuf[m * CFP + floc] = p;  // ds_store_b32, contiguous across lanes
      }
    }
    __syncthreads();

    // ---- phase 2: WMMA over this K chunk ----
    for (int kb = 0; kb < CF; kb += 32) {
      // A fragments (contiguous K order, matching B):
      // lanes 0-15: K = kb+0..15 ; lanes 16-31: K = kb+16..31
      v16h a0, a1;
      {
        union { v16h v; v8h h[2]; } u;
        int base = nlane * CFP + kb + lhalf * 16;    // affine in kb
        u.h[0] = *(const v8h*)&hbuf[base];
        u.h[1] = *(const v8h*)&hbuf[base + 8];
        a0 = u.v;
        base += 16 * CFP;                            // rows 16-31
        u.h[0] = *(const v8h*)&hbuf[base];
        u.h[1] = *(const v8h*)&hbuf[base + 8];
        a1 = u.v;
      }
      // B fragment: lane = N, lanes 0-15 hold K 0-15, lanes 16-31 hold K 16-31
      int kglob = ch * CF + kb + lhalf * 16;
#pragma unroll
      for (int t = 0; t < 4; ++t) {
        int col = wave * 64 + t * 16 + nlane;
        v16h bf = *(const v16h*)&w2h[(size_t)col * FFN + kglob];  // 32B contiguous
        acc0[t] = __builtin_amdgcn_wmma_f32_16x16x32_f16(
            false, a0, false, bf, (short)0, acc0[t], false, false);
        acc1[t] = __builtin_amdgcn_wmma_f32_16x16x32_f16(
            false, a1, false, bf, (short)0, acc1[t], false, false);
      }
    }
  }

  // ---- epilogue: + b2, store F32 per C/D layout (VGPR r -> M = r + 8*lhalf) ----
#pragma unroll
  for (int t = 0; t < 4; ++t) {
    int col = wave * 64 + t * 16 + nlane;
    float bias = b2[col];
#pragma unroll
    for (int r = 0; r < 8; ++r) {
      int m0 = r + lhalf * 8;
      out[(size_t)(rowbase + m0) * EMBED + col]      = acc0[t][r] + bias;
      out[(size_t)(rowbase + 16 + m0) * EMBED + col] = acc1[t][r] + bias;
    }
  }
}

extern "C" void kernel_launch(void* const* d_in, const int* in_sizes, int n_in,
                              void* d_out, int out_size, void* d_ws, size_t ws_size,
                              hipStream_t stream) {
  const float* x     = (const float*)d_in[0];
  const float* theta = (const float*)d_in[1];
  const float* W1    = (const float*)d_in[2];
  const float* b1    = (const float*)d_in[3];
  const float* W2    = (const float*)d_in[4];
  const float* b2    = (const float*)d_in[5];
  float* out = (float*)d_out;

  _Float16* w2h = (_Float16*)d_ws;  // 512*2048*2 = 2 MB scratch

  // Convert W2 to fp16 (1,048,576 elems, 4 per thread)
  w2_cvt<<<(EMBED * FFN) / (256 * 4), 256, 0, stream>>>(W2, w2h);

  // Fused kernel: 65536 rows / 32 per WG = 2048 workgroups of 256 threads
  qffn<<<(32 * 2048) / MT, 256, 0, stream>>>(x, theta, W1, b1, w2h, b2, out);
}
